// Attention_85177791414349
// MI455X (gfx1250) — compile-verified, matrix operand order checked by reading, not run on hardware
//
#include <hip/hip_runtime.h>
#include <hip/hip_bf16.h>

typedef __attribute__((ext_vector_type(16))) __bf16 v16bf;
typedef __attribute__((ext_vector_type(8)))  __bf16 bf16x8;
typedef __attribute__((ext_vector_type(8)))  float  v8f;

#define FA_NEG_INF (-1e30f)

namespace {
constexpr int kB      = 8;
constexpr int kTq     = 2048;
constexpr int kTk     = 2048;
constexpr int kD      = 512;
constexpr int kWaves  = 4;              // waves per workgroup
constexpr int kDSlice = kD / kWaves;    // 128 dims per wave
constexpr int kQT     = 16;             // query rows per workgroup
constexpr int kKT     = 64;             // keys per step (4 x 16-key subtiles)
}

__device__ __forceinline__ v8f wmma_bf16(v16bf a, v16bf b, v8f c) {
  // D = A(16x32 bf16) * B(32x16 bf16) + C(16x16 f32)
  return __builtin_amdgcn_wmma_f32_16x16x32_bf16(false, a, false, b,
                                                 (short)0, c, false, false);
}

// DPP-based lane shuffle (stays within each 16-lane half for these ctrls).
template <int CTRL>
__device__ __forceinline__ float dpp_xor(float x) {
  int xi = __float_as_int(x);
  int yi = __builtin_amdgcn_update_dpp(xi, xi, CTRL, 0xf, 0xf, true);
  return __int_as_float(yi);
}
// Butterfly over 16 lanes with independent xor masks {1,2,7,15}:
// quad_perm[1,0,3,2]=0xB1 (xor1), quad_perm[2,3,0,1]=0x4E (xor2),
// row_half_mirror=0x141 (xor7), row_mirror=0x140 (xor15).
__device__ __forceinline__ float row_max16(float x) {
  x = fmaxf(x, dpp_xor<0xB1>(x));
  x = fmaxf(x, dpp_xor<0x4E>(x));
  x = fmaxf(x, dpp_xor<0x141>(x));
  x = fmaxf(x, dpp_xor<0x140>(x));
  return x;
}
__device__ __forceinline__ float row_sum16(float x) {
  x += dpp_xor<0xB1>(x);
  x += dpp_xor<0x4E>(x);
  x += dpp_xor<0x141>(x);
  x += dpp_xor<0x140>(x);
  return x;
}

// Build a 16x32 A-fragment (or the symmetric B=K^T fragment) from an fp32
// row-major row pointer. Lane layout: lanes 0-15 hold K runs {k0..k0+7,
// k0+16..k0+23}; lanes 16-31 hold {k0+8..k0+15, k0+24..k0+31}.
__device__ __forceinline__ v16bf load_frag_bf16(const float* __restrict__ rowp,
                                                int k0, int h) {
  const float* p0 = rowp + k0 + 8 * h;
  const float* p1 = p0 + 16;
  float4 a = *(const float4*)(p0);
  float4 b = *(const float4*)(p0 + 4);
  float4 c = *(const float4*)(p1);
  float4 d = *(const float4*)(p1 + 4);
  v16bf r;
  r[0]  = (__bf16)a.x; r[1]  = (__bf16)a.y; r[2]  = (__bf16)a.z; r[3]  = (__bf16)a.w;
  r[4]  = (__bf16)b.x; r[5]  = (__bf16)b.y; r[6]  = (__bf16)b.z; r[7]  = (__bf16)b.w;
  r[8]  = (__bf16)c.x; r[9]  = (__bf16)c.y; r[10] = (__bf16)c.z; r[11] = (__bf16)c.w;
  r[12] = (__bf16)d.x; r[13] = (__bf16)d.y; r[14] = (__bf16)d.z; r[15] = (__bf16)d.w;
  return r;
}

__global__ __launch_bounds__(kWaves * 32, 1)
void fa_fwd_f32_bf16wmma(const float* __restrict__ Q, const float* __restrict__ K,
                         const float* __restrict__ V, const int* __restrict__ AM,
                         float* __restrict__ O) {
  // 88KB static LDS (within CDNA5's 320KB/WG limit)
  __shared__ __attribute__((aligned(16))) float  sred[kWaves][32][kKT / 2]; // 16KB
  __shared__ __attribute__((aligned(16))) __bf16 pbuf[kWaves][kQT][kKT];    // 8KB
  __shared__ __attribute__((aligned(16))) __bf16 vt[kD][kKT];               // 64KB

  const int qb   = blockIdx.x * kQT;
  const int b    = blockIdx.y;
  const int tid  = (int)threadIdx.x;
  const int lane = tid & 31;
  const int wave = tid >> 5;
  const int h    = lane >> 4;   // half-wave
  const int lq   = lane & 15;

  const float* Qb = Q + (size_t)b * kTq * kD;
  const float* Kb = K + (size_t)b * kTk * kD;
  const float* Vb = V + (size_t)b * kTk * kD;
  const int*   Mb = AM + (size_t)b * kTk;
  float*       Ob = O + (size_t)b * kTq * kD;

  // ---- Q A-fragments: loaded once, held in VGPRs (4 x v16bf = 32 VGPRs) ----
  v16bf qf[4];
  {
    const float* qrow = Qb + (size_t)(qb + lq) * kD;
#pragma unroll
    for (int c = 0; c < 4; ++c)
      qf[c] = load_frag_bf16(qrow, wave * kDSlice + 32 * c, h);
  }

  v8f acc[8];
#pragma unroll
  for (int j = 0; j < 8; ++j)
#pragma unroll
    for (int i = 0; i < 8; ++i) acc[j][i] = 0.0f;

  float mrow[8], lrow[8];   // running max (base-2 domain) and denominator
#pragma unroll
  for (int r = 0; r < 8; ++r) { mrow[r] = FA_NEG_INF; lrow[r] = 0.0f; }

  // softmax(s/sqrt(D)) computed as exp2((s * scale2) - m2); scale2 folds log2(e)
  const float scale2 = 0.044194173824159216f * 1.4426950408889634f;
  const int kend = qb + kQT;                  // causal: keys 0..qb+15 needed

  for (int kb = 0; kb < kend; kb += kKT) {
    // ---- stage V tile (64 keys x 512 dims fp32) transposed into LDS as bf16 ----
    // Each thread owns 4 dims (tid*4..tid*4+3); per group of 8 keys it loads
    // 8 float4s and emits 4 ds_store_b128 (8 packed bf16 keys per dim).
#pragma unroll
    for (int g = 0; g < kKT / 8; ++g) {
      const float* vp = Vb + (size_t)(kb + g * 8) * kD + tid * 4;
      float lv[8][4];
#pragma unroll
      for (int j = 0; j < 8; ++j) {
        float4 t = *(const float4*)(vp + (size_t)j * kD);
        lv[j][0] = t.x; lv[j][1] = t.y; lv[j][2] = t.z; lv[j][3] = t.w;
      }
#pragma unroll
      for (int i = 0; i < 4; ++i) {
        union { __bf16 hh[8]; uint4 u; } pk;
#pragma unroll
        for (int j = 0; j < 8; ++j) pk.hh[j] = (__bf16)lv[j][i];
        *(uint4*)&vt[tid * 4 + i][g * 8] = pk.u;
      }
    }

    if (kb + kKT < kend) {  // prefetch next tiles (global_prefetch_b8)
      __builtin_prefetch(Vb + (size_t)(kb + kKT) * kD + tid * 4, 0, 0);
      __builtin_prefetch(Kb + (size_t)(kb + kKT) * kD + tid * 4, 0, 0);
    }

    // ---- partial S = Q_slice * K_slice^T over this wave's 128 dims ----
    v8f s[4];
#pragma unroll
    for (int t = 0; t < 4; ++t)
#pragma unroll
      for (int i = 0; i < 8; ++i) s[t][i] = 0.0f;
#pragma unroll
    for (int t = 0; t < 4; ++t) {
      const float* krow = Kb + (size_t)(kb + 16 * t + lq) * kD;
#pragma unroll
      for (int c = 0; c < 4; ++c)
        s[t] = wmma_bf16(qf[c], load_frag_bf16(krow, wave * kDSlice + 32 * c, h), s[t]);
    }

    // ---- lane-major cross-wave reduction of S through LDS ----
    {
      float4* sp = (float4*)&sred[wave][lane][0];
#pragma unroll
      for (int t = 0; t < 4; ++t) {
        sp[2 * t]     = make_float4(s[t][0], s[t][1], s[t][2], s[t][3]);
        sp[2 * t + 1] = make_float4(s[t][4], s[t][5], s[t][6], s[t][7]);
      }
    }
    __syncthreads();
#pragma unroll
    for (int ow = 0; ow < kWaves; ++ow) {
      if (ow == wave) continue;
      const float4* op = (const float4*)&sred[ow][lane][0];
#pragma unroll
      for (int t = 0; t < 4; ++t) {
        float4 x0 = op[2 * t], x1 = op[2 * t + 1];
        s[t][0] += x0.x; s[t][1] += x0.y; s[t][2] += x0.z; s[t][3] += x0.w;
        s[t][4] += x1.x; s[t][5] += x1.y; s[t][6] += x1.z; s[t][7] += x1.w;
      }
    }

    // ---- scale + causal/padding mask (C-layout: lane holds col, 8 rows) ----
    int keyc[4], amc[4];
#pragma unroll
    for (int t = 0; t < 4; ++t) {
      keyc[t] = kb + 16 * t + lq;
      amc[t]  = (keyc[t] < kTk) ? Mb[keyc[t]] : 0;
    }
#pragma unroll
    for (int r = 0; r < 8; ++r) {
      const int qr = qb + r + 8 * h;
#pragma unroll
      for (int t = 0; t < 4; ++t)
        s[t][r] = (amc[t] && keyc[t] <= qr) ? s[t][r] * scale2 : FA_NEG_INF;
    }

    // ---- online softmax (base-2); row reductions via DPP butterflies ----
#pragma unroll
    for (int r = 0; r < 8; ++r) {
      float mx = fmaxf(fmaxf(s[0][r], s[1][r]), fmaxf(s[2][r], s[3][r]));
      mx = row_max16(mx);
      const float mn   = fmaxf(mrow[r], mx);
      const float corr = exp2f(mrow[r] - mn);
      float p[4];
#pragma unroll
      for (int t = 0; t < 4; ++t) p[t] = exp2f(s[t][r] - mn);
      const float sum = row_sum16((p[0] + p[1]) + (p[2] + p[3]));
      lrow[r] = lrow[r] * corr + sum;
      mrow[r] = mn;
#pragma unroll
      for (int j = 0; j < 8; ++j) acc[j][r] *= corr;
      const int prow = r + 8 * h;                 // C-layout -> LDS (row-major)
#pragma unroll
      for (int t = 0; t < 4; ++t)
        pbuf[wave][prow][16 * t + lq] = (__bf16)p[t];
    }
    __syncthreads();

    // ---- O += P(16x64) * V(64x16) per owned column tile (2 K=32 chunks) ----
    v16bf pf[2];
#pragma unroll
    for (int c = 0; c < 2; ++c) {
      bf16x8 pa = *(const bf16x8*)&pbuf[wave][lq][32 * c + 8 * h];
      bf16x8 pb = *(const bf16x8*)&pbuf[wave][lq][32 * c + 16 + 8 * h];
#pragma unroll
      for (int i = 0; i < 8; ++i) { pf[c][i] = pa[i]; pf[c][8 + i] = pb[i]; }
    }
#pragma unroll
    for (int j = 0; j < 8; ++j) {
      const int d0 = wave * kDSlice + 16 * j;
#pragma unroll
      for (int c = 0; c < 2; ++c) {
        bf16x8 va = *(const bf16x8*)&vt[d0 + lq][32 * c + 8 * h];
        bf16x8 vb = *(const bf16x8*)&vt[d0 + lq][32 * c + 16 + 8 * h];
        v16bf vf;
#pragma unroll
        for (int i = 0; i < 8; ++i) { vf[i] = va[i]; vf[8 + i] = vb[i]; }
        acc[j] = wmma_bf16(pf[c], vf, acc[j]);
      }
    }
    __syncthreads();   // protects vt/sred/pbuf for next iteration
  }

  // ---- epilogue: normalize and store ----
#pragma unroll
  for (int j = 0; j < 8; ++j) {
    const int col = wave * kDSlice + 16 * j + lq;
#pragma unroll
    for (int r = 0; r < 8; ++r) {
      const int row = qb + r + 8 * h;
      Ob[(size_t)row * kD + col] = acc[j][r] / lrow[r];
    }
  }
}

extern "C" void kernel_launch(void* const* d_in, const int* in_sizes, int n_in,
                              void* d_out, int out_size, void* d_ws, size_t ws_size,
                              hipStream_t stream) {
  (void)in_sizes; (void)n_in; (void)out_size; (void)d_ws; (void)ws_size;
  const float* Q  = (const float*)d_in[0];
  const float* K  = (const float*)d_in[1];
  const float* V  = (const float*)d_in[2];
  const int*   AM = (const int*)d_in[3];
  float*       O  = (float*)d_out;

  dim3 grid(kTq / kQT, kB);   // 128 query tiles x 8 batches
  dim3 block(kWaves * 32);    // 4 wave32s per workgroup
  hipLaunchKernelGGL(fa_fwd_f32_bf16wmma, grid, block, 0, stream, Q, K, V, AM, O);
}
